// AltAttention_76295799046479
// MI455X (gfx1250) — compile-verified, matrix operand order checked by reading, not run on hardware
//
#include <hip/hip_runtime.h>
#include <hip/hip_bf16.h>
#include <math.h>

typedef __bf16 bf16_t;
typedef __attribute__((ext_vector_type(16))) __bf16        v16bf;
typedef __attribute__((ext_vector_type(8)))  __bf16        v8bf;
typedef __attribute__((ext_vector_type(8)))  float         v8f;
typedef __attribute__((ext_vector_type(4)))  unsigned int  v4u;
typedef __attribute__((ext_vector_type(8)))  int           v8i;
typedef __attribute__((ext_vector_type(4)))  int           v4i;

#define DIMN   1024
#define SLEN   2048
#define BATCH  2
#define NHEAD  16
#define HD     64
#define QKVW   (3 * DIMN)          // 3072
#define SCALE  0.03125f            // 1024^-0.5 (reference scales by full dim)

// LDS tile row stride in halves: 64 data halves (32 DW) + TDM pad of 4 DW
#define KVSTR  72
// TDM pad config: pad_enable, pad_interval code 4 (=32 DWORDs),
// pad_amount code 3 (=4 DWORDs)  ->  row stride 36 DW = 72 halves
#define PADCFG ((1u << 20) | (4u << 22) | (3u << 25))

// ---------------------------------------------------------------------------
// TDM: issue a 2D tensor tile load into LDS (D# built per ISA ch.8).
// gaddr points at the TILE start inside the tensor. All units = 2-byte elems.
// This toolchain exposes the 6-arg builtin:
//   (uint32x4 g0, int32x8 g1, int32x4 g2, int32x4 g3, int32x8 extra, i32 cpol)
// ---------------------------------------------------------------------------
__device__ __forceinline__ unsigned lds_off_u32(const void* p) {
  return (unsigned)(uintptr_t)p;   // low 32 bits of flat LDS addr = LDS offset
}

__device__ __forceinline__ void tdm_load_2d(const void* gaddr, unsigned ldsoff,
                                            unsigned td0, unsigned td1,
                                            unsigned long long stride0,
                                            unsigned tile_d0, unsigned tile_d1) {
  const unsigned long long ga = (unsigned long long)(uintptr_t)gaddr;
  v4u g0;
  g0.x = 1u;                                   // count=1 (valid user D#)
  g0.y = ldsoff;                               // lds_addr
  g0.z = (unsigned)ga;                         // global_addr[31:0]
  g0.w = (unsigned)(ga >> 32) | (2u << 30);    // global_addr[56:32] | type=2
  v8i g1;
  g1[0] = (int)((1u << 16) | PADCFG);          // data_size=1 (2B) + padding
  g1[1] = (int)((td0 & 0xffffu) << 16);                                  // dim0 lo
  g1[2] = (int)(((td0 >> 16) & 0xffffu) | ((td1 & 0xffffu) << 16));      // dim0 hi | dim1 lo
  g1[3] = (int)(((td1 >> 16) & 0xffffu) | ((tile_d0 & 0xffffu) << 16));  // dim1 hi | tile_dim0
  g1[4] = (int)(tile_d1 & 0xffffu);            // tile_dim1 (tile_dim2 = 0)
  g1[5] = (int)(unsigned)stride0;              // tensor_dim0_stride lo32
  g1[6] = (int)((unsigned)(stride0 >> 32) & 0xffffu);  // stride hi16
  g1[7] = 0;
  const v4i z4 = {0, 0, 0, 0};
  const v8i z8 = {0, 0, 0, 0, 0, 0, 0, 0};
  __builtin_amdgcn_tensor_load_to_lds(g0, g1, z4, z4, z8, 0);
}

// ---------------------------------------------------------------------------
// Fragment loaders matching the CDNA5 16-bit A-matrix (16x32) VGPR layout:
//   lanes 0-15 : row=row0+lane,    halves 0-7 = K[k0..k0+7],  8-15 = K[k0+16..23]
//   lanes 16-31: row=row0+lane-16, halves 0-7 = K[k0+8..15],  8-15 = K[k0+24..31]
// Same pattern serves A-fragments and B-fragments (columns stored as rows).
// ---------------------------------------------------------------------------
__device__ __forceinline__ v16bf load_frag(const bf16_t* __restrict__ base, int ld,
                                           int row0, int col0, int lane) {
  const int r  = row0 + (lane & 15);
  const int kb = col0 + ((lane >> 4) << 3);
  const bf16_t* p = base + (size_t)r * (size_t)ld + kb;
  v8bf lo = *(const v8bf*)(p);
  v8bf hi = *(const v8bf*)(p + 16);
  v16bf a;
#pragma unroll
  for (int i = 0; i < 8; ++i) { a[i] = lo[i]; a[i + 8] = hi[i]; }
  return a;
}

__device__ __forceinline__ v16bf lds_frag(const bf16_t* base, int row0, int col0,
                                          int lane, int stride) {
  const bf16_t* p = base + (row0 + (lane & 15)) * stride + col0 + ((lane >> 4) << 3);
  v8bf lo = *(const v8bf*)(p);
  v8bf hi = *(const v8bf*)(p + 16);
  v16bf a;
#pragma unroll
  for (int i = 0; i < 8; ++i) { a[i] = lo[i]; a[i + 8] = hi[i]; }
  return a;
}

__device__ __forceinline__ v8f wmma_bf16(v16bf a, v16bf b, v8f c) {
  return __builtin_amdgcn_wmma_f32_16x16x32_bf16(false, a, false, b,
                                                 (short)0, c, false, false);
}

// ---------------------------------------------------------------------------
// Elementwise converters
// ---------------------------------------------------------------------------
__global__ __launch_bounds__(256)
void cvt_f32_bf16(const float* __restrict__ x, bf16_t* __restrict__ y, int n) {
  for (int i = blockIdx.x * blockDim.x + threadIdx.x; i < n;
       i += gridDim.x * blockDim.x)
    y[i] = (bf16_t)x[i];
}

// W[K,N] f32 -> Wt[N,K] bf16
__global__ __launch_bounds__(256)
void cvt_transpose_bf16(const float* __restrict__ w, bf16_t* __restrict__ wt,
                        int K, int N) {
  const int n = K * N;
  for (int i = blockIdx.x * blockDim.x + threadIdx.x; i < n;
       i += gridDim.x * blockDim.x) {
    const int nr = i / K;
    const int kc = i - nr * K;
    wt[i] = (bf16_t)w[(size_t)kc * N + nr];
  }
}

// qkvb [B*S, 3D] bf16 -> Vt [(b,h), HD, S] bf16 (per-head V transposed)
__global__ __launch_bounds__(256)
void v_transpose(const bf16_t* __restrict__ qkvb, bf16_t* __restrict__ vt) {
  const int total = BATCH * NHEAD * HD * SLEN;
  for (int i = blockIdx.x * blockDim.x + threadIdx.x; i < total;
       i += gridDim.x * blockDim.x) {
    const int s  = i & (SLEN - 1);
    const int d  = (i >> 11) & (HD - 1);
    const int bh = i >> 17;
    const int b  = bh >> 4;
    const int h  = bh & 15;
    vt[i] = qkvb[(size_t)(b * SLEN + s) * QKVW + h * (3 * HD) + 2 * HD + d];
  }
}

// ---------------------------------------------------------------------------
// GEMM:  C[M,N] = A[M,K](bf16) @ Bt[N,K]^T(bf16) + bias[N]
// Block = 8 waves -> 128x64 C tile. Per 64-wide K step, wave 0 TDM-loads the
// shared A tile (128x64) and B tile (64x64) into padded LDS; every wave then
// runs 8 WMMAs off ds_load_b128 fragments. Grid = (M/128, N/64).
// ---------------------------------------------------------------------------
template <bool OUT_BF16>
__global__ __launch_bounds__(256)
void gemm_bias(const bf16_t* __restrict__ A, const bf16_t* __restrict__ Bt,
               const float* __restrict__ bias, void* __restrict__ Cout,
               int M, int N, int K) {
  __shared__ __attribute__((aligned(16))) bf16_t smem[(128 + 64) * KVSTR];
  bf16_t* ldsA = smem;                 // 128 rows x 72 halves
  bf16_t* ldsB = smem + 128 * KVSTR;   //  64 rows x 72 halves

  const int lane  = threadIdx.x & 31;
  const int wave  = threadIdx.x >> 5;
  const int rowB0 = blockIdx.x * 128;          // block row base
  const int row0  = wave * 16;                 // wave row inside A tile
  const int col0  = blockIdx.y * 64;

  v8f acc[4];
#pragma unroll
  for (int t = 0; t < 4; ++t)
#pragma unroll
    for (int r = 0; r < 8; ++r) acc[t][r] = 0.0f;

  for (int kt = 0; kt < K; kt += 64) {
    __syncthreads();                            // previous tile fully consumed
    if (wave == 0) {
      tdm_load_2d(A  + (size_t)rowB0 * K + kt, lds_off_u32(ldsA),
                  (unsigned)K, (unsigned)M, (unsigned long long)K, 64u, 128u);
      tdm_load_2d(Bt + (size_t)col0 * K + kt, lds_off_u32(ldsB),
                  (unsigned)K, (unsigned)N, (unsigned long long)K, 64u, 64u);
      __builtin_amdgcn_s_wait_tensorcnt(0);
    }
    __syncthreads();                            // tiles visible to all waves

#pragma unroll
    for (int sub = 0; sub < 64; sub += 32) {
      v16bf a = lds_frag(ldsA, row0, sub, lane, KVSTR);
#pragma unroll
      for (int t = 0; t < 4; ++t) {
        v16bf b = lds_frag(ldsB, t * 16, sub, lane, KVSTR);
        acc[t] = wmma_bf16(a, b, acc[t]);
      }
    }
  }

  const int cn  = lane & 15;
  const int rhi = (lane >> 4) << 3;
#pragma unroll
  for (int t = 0; t < 4; ++t) {
    const int col = col0 + t * 16 + cn;
    const float bv = bias[col];
#pragma unroll
    for (int r = 0; r < 8; ++r) {
      const int row = rowB0 + row0 + r + rhi;
      const float v = acc[t][r] + bv;
      if (OUT_BF16)
        ((bf16_t*)Cout)[(size_t)row * N + col] = (bf16_t)v;
      else
        ((float*)Cout)[(size_t)row * N + col] = v;
    }
  }
}

// ---------------------------------------------------------------------------
// Flash attention: grid (B*H, S/128), block 256 (8 waves, 16 queries/wave).
// 64-key tiles; K/V tiles TDM-staged to LDS (shared by all 8 waves).
// Row-sum of P computed with WMMA against an all-ones fragment (no shuffles);
// only the row-max uses width-16 xor reductions.
// ---------------------------------------------------------------------------
__global__ __launch_bounds__(256)
void flash_attn(const bf16_t* __restrict__ qkvb, const bf16_t* __restrict__ vt,
                bf16_t* __restrict__ outb) {
  __shared__ __attribute__((aligned(16))) bf16_t smem[(64 + 64 + 8 * 16) * KVSTR];
  bf16_t* ldsK = smem;                          // 64 x 72
  bf16_t* ldsV = smem + 64 * KVSTR;             // 64 x 72
  bf16_t* ldsP = smem + 128 * KVSTR;            // 8 waves x 16 x 72

  const int lane = threadIdx.x & 31;
  const int wave = threadIdx.x >> 5;
  const int bh   = blockIdx.x;
  const int b    = bh >> 4;
  const int h    = bh & 15;
  const int q0   = blockIdx.y * 128 + wave * 16;
  const int rowQ = b * SLEN + q0;
  const int cq   = h * (3 * HD);
  const int ck   = cq + HD;
  const int vrow = bh * HD;
  bf16_t* myP = ldsP + wave * 16 * KVSTR;

  // Q fragments held in registers for the whole key loop (K-dim 64 = 2 steps)
  const v16bf aQ0 = load_frag(qkvb, QKVW, rowQ, cq + 0,  lane);
  const v16bf aQ1 = load_frag(qkvb, QKVW, rowQ, cq + 32, lane);

  v16bf bOnes;
#pragma unroll
  for (int i = 0; i < 16; ++i) bOnes[i] = (bf16_t)1.0f;

  v8f o[4];
  float mrow[8], lrow[8];
#pragma unroll
  for (int r = 0; r < 8; ++r) {
    o[0][r] = o[1][r] = o[2][r] = o[3][r] = 0.0f;
    mrow[r] = -3.0e38f;
    lrow[r] = 0.0f;
  }

  for (int j0 = 0; j0 < SLEN; j0 += 64) {
    __syncthreads();                            // previous K/V tile consumed
    if (wave == 0) {
      tdm_load_2d(qkvb + (size_t)(b * SLEN + j0) * QKVW + ck, lds_off_u32(ldsK),
                  (unsigned)QKVW, (unsigned)(BATCH * SLEN),
                  (unsigned long long)QKVW, 64u, 64u);
      tdm_load_2d(vt + (size_t)vrow * SLEN + j0, lds_off_u32(ldsV),
                  (unsigned)SLEN, (unsigned)(BATCH * NHEAD * HD),
                  (unsigned long long)SLEN, 64u, 64u);
      __builtin_amdgcn_s_wait_tensorcnt(0);
    }
    __syncthreads();

    // ---- scores: S[16q x 64k] = Q.K^T  (8 WMMAs) -----------------------
    v8f s[4];
#pragma unroll
    for (int c = 0; c < 4; ++c) {
#pragma unroll
      for (int r = 0; r < 8; ++r) s[c][r] = 0.0f;
      v16bf bK0 = lds_frag(ldsK, c * 16, 0,  lane, KVSTR);
      v16bf bK1 = lds_frag(ldsK, c * 16, 32, lane, KVSTR);
      s[c] = wmma_bf16(aQ0, bK0, s[c]);
      s[c] = wmma_bf16(aQ1, bK1, s[c]);
    }

    // ---- online softmax: row-max via width-16 reductions ---------------
#pragma unroll
    for (int r = 0; r < 8; ++r) {
      const float x0 = s[0][r] * SCALE;
      const float x1 = s[1][r] * SCALE;
      const float x2 = s[2][r] * SCALE;
      const float x3 = s[3][r] * SCALE;
      float tm = fmaxf(fmaxf(x0, x1), fmaxf(x2, x3));
#pragma unroll
      for (int off = 8; off >= 1; off >>= 1)
        tm = fmaxf(tm, __shfl_xor(tm, off, 32));
      const float mnew  = fmaxf(mrow[r], tm);
      const float alpha = __expf(mrow[r] - mnew);
      const float p0 = __expf(x0 - mnew);
      const float p1 = __expf(x1 - mnew);
      const float p2 = __expf(x2 - mnew);
      const float p3 = __expf(x3 - mnew);
      mrow[r] = mnew;
      lrow[r] *= alpha;
      o[0][r] *= alpha; o[1][r] *= alpha; o[2][r] *= alpha; o[3][r] *= alpha;

      const int row = r + ((lane >> 4) << 3);   // C-layout row of this elem
      const int col = lane & 15;
      myP[row * KVSTR + col]      = (bf16_t)p0;
      myP[row * KVSTR + col + 16] = (bf16_t)p1;
      myP[row * KVSTR + col + 32] = (bf16_t)p2;
      myP[row * KVSTR + col + 48] = (bf16_t)p3;
    }
    asm volatile("s_wait_dscnt 0" ::: "memory");   // wave-private LDS RAW

    // ---- reload P as two 16x32 A-fragments -----------------------------
    const v16bf aP0 = lds_frag(myP, 0, 0,  lane, KVSTR);
    const v16bf aP1 = lds_frag(myP, 0, 32, lane, KVSTR);

    // ---- row-sum(P) = P . ones  (2 WMMAs, lands in row-stat layout) ----
    v8f lt;
#pragma unroll
    for (int r = 0; r < 8; ++r) lt[r] = 0.0f;
    lt = wmma_bf16(aP0, bOnes, lt);
    lt = wmma_bf16(aP1, bOnes, lt);
#pragma unroll
    for (int r = 0; r < 8; ++r) lrow[r] += lt[r];

    // ---- O += P @ V  (8 WMMAs) -----------------------------------------
#pragma unroll
    for (int t = 0; t < 4; ++t) {
      o[t] = wmma_bf16(aP0, lds_frag(ldsV, t * 16, 0,  lane, KVSTR), o[t]);
      o[t] = wmma_bf16(aP1, lds_frag(ldsV, t * 16, 32, lane, KVSTR), o[t]);
    }
  }

  // ---- normalize and emit bf16 [B*S, D] for the proj GEMM --------------
  const int cn  = lane & 15;
  const int rhi = (lane >> 4) << 3;
#pragma unroll
  for (int r = 0; r < 8; ++r) {
    const float inv = 1.0f / lrow[r];
    const int row   = b * SLEN + q0 + r + rhi;
    const size_t rb = (size_t)row * DIMN + h * HD;
    outb[rb + 0  + cn] = (bf16_t)(o[0][r] * inv);
    outb[rb + 16 + cn] = (bf16_t)(o[1][r] * inv);
    outb[rb + 32 + cn] = (bf16_t)(o[2][r] * inv);
    outb[rb + 48 + cn] = (bf16_t)(o[3][r] * inv);
  }
}

// ---------------------------------------------------------------------------
// Host-side orchestration
// ---------------------------------------------------------------------------
extern "C" void kernel_launch(void* const* d_in, const int* in_sizes, int n_in,
                              void* d_out, int out_size, void* d_ws, size_t ws_size,
                              hipStream_t stream) {
  const float* x      = (const float*)d_in[0];   // [B,S,D]
  const float* w_qkv  = (const float*)d_in[1];   // [D,3D]
  const float* b_qkv  = (const float*)d_in[2];   // [3D]
  const float* w_proj = (const float*)d_in[3];   // [D,D]
  const float* b_proj = (const float*)d_in[4];   // [D]
  float* out = (float*)d_out;                    // [B,S,D] f32

  char* ws = (char*)d_ws;
  const int MROWS = BATCH * SLEN;                // 4096
  size_t off = 0;
  bf16_t* xb    = (bf16_t*)(ws + off); off += (size_t)MROWS * DIMN * 2;      // 8 MB
  bf16_t* wqkvT = (bf16_t*)(ws + off); off += (size_t)QKVW * DIMN * 2;       // 6 MB
  bf16_t* wprjT = (bf16_t*)(ws + off); off += (size_t)DIMN * DIMN * 2;       // 2 MB
  bf16_t* qkvb  = (bf16_t*)(ws + off); off += (size_t)MROWS * QKVW * 2;      // 24 MB
  bf16_t* vtb   = (bf16_t*)(ws + off); off += (size_t)BATCH * NHEAD * HD * SLEN * 2; // 8 MB
  bf16_t* attnb = (bf16_t*)(ws + off); off += (size_t)MROWS * DIMN * 2;      // 8 MB

  // 1) down-convert activations & weights (weights transposed to [N,K])
  cvt_f32_bf16<<<1024, 256, 0, stream>>>(x, xb, MROWS * DIMN);
  cvt_transpose_bf16<<<4096, 256, 0, stream>>>(w_qkv, wqkvT, DIMN, QKVW);
  cvt_transpose_bf16<<<2048, 256, 0, stream>>>(w_proj, wprjT, DIMN, DIMN);

  // 2) QKV projection: qkvb = xb @ w_qkv + b_qkv  (bf16 out)
  gemm_bias<true><<<dim3(MROWS / 128, QKVW / 64), 256, 0, stream>>>(
      xb, wqkvT, b_qkv, qkvb, MROWS, QKVW, DIMN);

  // 3) per-head V transpose for contiguous P.V operands
  v_transpose<<<4096, 256, 0, stream>>>(qkvb, vtb);

  // 4) flash attention -> attnb (bf16, [B*S, D])
  flash_attn<<<dim3(BATCH * NHEAD, SLEN / 128), 256, 0, stream>>>(qkvb, vtb, attnb);

  // 5) output projection: out = attnb @ w_proj + b_proj (f32 out)
  gemm_bias<false><<<dim3(MROWS / 128, DIMN / 64), 256, 0, stream>>>(
      attnb, wprjT, b_proj, out, MROWS, DIMN, DIMN);
}
